// AtomAttention_46703474376855
// MI455X (gfx1250) — matrix operand order repaired and checked by reading, not compiled, not run on hardware
//
#include <hip/hip_runtime.h>

// ---------------------------------------------------------------------------
// AtomAttention for MI455X (gfx1250): block-diagonal attention, 2048 x (64x512)
// f16-input WMMA (v_wmma_f32_16x16x32_f16) with fp32 accumulation throughout.
// 512 threads (16 wave32) per block for better WMMA latency / barrier hiding.
// ---------------------------------------------------------------------------

typedef __attribute__((ext_vector_type(16))) _Float16 v16h;
typedef __attribute__((ext_vector_type(8)))  _Float16 v8h;
typedef __attribute__((ext_vector_type(4)))  _Float16 v4h;
typedef __attribute__((ext_vector_type(8)))  float    v8f;

#define NB   2048
#define MM   64
#define DD   512
#define NT   512            // threads per block (16 waves)
#define XPAD 520            // row stride (halves) for 64x512 f16 arrays in LDS
#define VPAD 72             // row stride (halves) for vhT / probs
#define SCPAD 68            // row stride (floats) for scores

// LDS byte offsets (dynamic shared memory)
#define XH_OFF 0                       // x (f16)   64 x XPAD   -> 66560 B (reused)
#define SC_OFF 0                       // scores (f32) 64 x SCPAD (overlays xh)
#define PH_OFF 17408                   // probs (f16) 64 x VPAD  (overlays xh)
#define QH_OFF 66560                   // Q (f16)   64 x XPAD
#define KH_OFF 133120                  // K (f16)   64 x XPAD
#define VT_OFF 199680                  // V^T (f16) 512 x VPAD  -> 73728 B
#define SMEM_BYTES (VT_OFF + DD * VPAD * 2)   // 273408 B <= 320 KB

// Build a v16h A/B fragment from two 16-byte chunks.
__device__ __forceinline__ v16h ld16(const _Float16* p0, const _Float16* p1) {
    v8h lo = *(const v8h*)p0;
    v8h hi = *(const v8h*)p1;
    v16h r;
#pragma unroll
    for (int i = 0; i < 8; ++i) { r[i] = lo[i]; r[i + 8] = hi[i]; }
    return r;
}

__device__ __forceinline__ v8f wmma_f16(v16h a, v16h b, v8f c) {
    return __builtin_amdgcn_wmma_f32_16x16x32_f16(false, a, false, b,
                                                  (short)0, c, false, false);
}

// ---------------------------------------------------------------------------
// Prep: Wq/Wk/Wv fp32 [d][e] -> wT f16 [p][e][d] (transposed, B-operand layout)
// ---------------------------------------------------------------------------
__global__ void __launch_bounds__(256) prep_weights(const float* __restrict__ Wq,
                                                    const float* __restrict__ Wk,
                                                    const float* __restrict__ Wv,
                                                    _Float16* __restrict__ wT) {
    int idx = blockIdx.x * 256 + threadIdx.x;      // 3*512*512 total
    if (idx >= 3 * DD * DD) return;
    int p   = idx >> 18;
    int rem = idx & (DD * DD - 1);
    int d   = rem >> 9;          // coalesced global read along e
    int e   = rem & (DD - 1);
    const float* W = (p == 0) ? Wq : (p == 1) ? Wk : Wv;
    wT[(p << 18) + e * DD + d] = (_Float16)W[d * DD + e];
}

// ---------------------------------------------------------------------------
// Main fused kernel: one block per molecule, 512 threads (16 waves).
// ---------------------------------------------------------------------------
__global__ void __launch_bounds__(NT, 1)
atom_attention_kernel(const float* __restrict__ x,
                      const int*   __restrict__ lengths,
                      const float* __restrict__ bq,
                      const float* __restrict__ bk,
                      const float* __restrict__ bv,
                      const _Float16* __restrict__ wT,
                      float* __restrict__ out) {
    extern __shared__ char smem[];
    _Float16* xh  = (_Float16*)(smem + XH_OFF);
    _Float16* qh  = (_Float16*)(smem + QH_OFF);
    _Float16* kh  = (_Float16*)(smem + KH_OFF);
    _Float16* vhT = (_Float16*)(smem + VT_OFF);
    float*    sc  = (float*)(smem + SC_OFF);
    _Float16* ph  = (_Float16*)(smem + PH_OFF);

    const int b     = blockIdx.x;
    const int tid   = threadIdx.x;
    const int wave  = tid >> 5;        // 0..15
    const int lane  = tid & 31;
    const int nlane = lane & 15;
    const int hi    = lane >> 4;       // 0 or 1
    const int hiA   = hi * 8;          // A-fragment K offset
    const int hi16  = hi * 16;         // unused; B fragments also use hiA
    const int len   = lengths[b];

    // ------------------- Phase 0: stage x[b] into LDS as f16 ---------------
    {
        const float4* xr = (const float4*)(x + (size_t)b * (MM * DD));
#pragma unroll
        for (int it = 0; it < 16; ++it) {
            int idx  = it * NT + tid;           // 8192 float4s
            float4 v = xr[idx];
            int flat = idx * 4;
            int row  = flat >> 9;
            int col  = flat & (DD - 1);
            v4h h = { (_Float16)v.x, (_Float16)v.y, (_Float16)v.z, (_Float16)v.w };
            *(v4h*)(xh + row * XPAD + col) = h;
        }
    }
    __syncthreads();

    // ------------------- Phase 1: Q/K/V = x*W + b --------------------------
    // 96 strips (3 proj x 32 n-tiles), 6 per wave; each strip = 4 M-tiles.
    for (int t = 0; t < 6; ++t) {
        int s = wave * 6 + t;
        int p = s >> 5;              // 0=Q 1=K 2=V
        int j = s & 31;              // n-tile
        const float* bias = (p == 0) ? bq : (p == 1) ? bk : bv;
        float bval = bias[j * 16 + nlane];
        v8f acc[4];
#pragma unroll
        for (int i = 0; i < 4; ++i)
#pragma unroll
            for (int r = 0; r < 8; ++r) acc[i][r] = bval;

        const _Float16* wrow = wT + ((size_t)p << 18) + (size_t)(j * 16 + nlane) * DD;
        for (int kb = 0; kb < 16; ++kb) {
            int koff = kb * 32;
            v16h bf = ld16(wrow + koff + hiA, wrow + koff + hiA + 16);
#pragma unroll
            for (int i = 0; i < 4; ++i) {
                const _Float16* arow = xh + (i * 16 + nlane) * XPAD + koff + hiA;
                v16h af = ld16(arow, arow + 16);
                acc[i] = wmma_f16(af, bf, acc[i]);
            }
        }
        // store: Q,K row-major; V transposed
#pragma unroll
        for (int i = 0; i < 4; ++i)
#pragma unroll
            for (int r = 0; r < 8; ++r) {
                int atom = i * 16 + r + hiA;
                _Float16 h = (_Float16)acc[i][r];
                if (p == 0)      qh[atom * XPAD + j * 16 + nlane] = h;
                else if (p == 1) kh[atom * XPAD + j * 16 + nlane] = h;
                else             vhT[(j * 16 + nlane) * VPAD + atom] = h;
            }
    }
    __syncthreads();

    // ------------------- Phase 2: scores = Q K^T * scale -------------------
    const float scale = 0.044194173824159216f;   // 1/sqrt(512)
    {
        int tile = wave;                          // 16 tiles, one per wave
        int qi = tile >> 2, ai = tile & 3;
        v8f acc = {};
        for (int kb = 0; kb < 16; ++kb) {
            int koff = kb * 32;
            const _Float16* arow = qh + (qi * 16 + nlane) * XPAD + koff + hiA;
            v16h af = ld16(arow, arow + 16);
            const _Float16* brow = kh + (ai * 16 + nlane) * XPAD + koff + hiA;
            v16h bf = ld16(brow, brow + 16);
            acc = wmma_f16(af, bf, acc);
        }
#pragma unroll
        for (int r = 0; r < 8; ++r)
            sc[(qi * 16 + r + hiA) * SCPAD + ai * 16 + nlane] = acc[r] * scale;
    }
    __syncthreads();

    // ------------------- Softmax per query row (masked keys) ---------------
    if (tid < MM) {
        int q = tid;
        if (q >= len) {
            for (int k = 0; k < MM; ++k) ph[q * VPAD + k] = (_Float16)0.0f;
        } else {
            float m = -3.0e38f;
            for (int k = 0; k < len; ++k) m = fmaxf(m, sc[q * SCPAD + k]);
            float ssum = 0.0f;
            for (int k = 0; k < len; ++k) {
                float e = __expf(sc[q * SCPAD + k] - m);
                sc[q * SCPAD + k] = e;
                ssum += e;
            }
            float inv = 1.0f / ssum;
            for (int k = 0; k < MM; ++k)
                ph[q * VPAD + k] = (k < len) ? (_Float16)(sc[q * SCPAD + k] * inv)
                                             : (_Float16)0.0f;
        }
    }
    __syncthreads();

    // ------------------- Phase 3: hid = P*V, masked mean -> out ------------
#pragma unroll
    for (int jj = 0; jj < 2; ++jj) {
        int j = wave * 2 + jj;                 // output-dim tile (0..31)
        float colsum = 0.0f;
#pragma unroll
        for (int i = 0; i < 4; ++i) {          // query tile
            v8f acc = {};
#pragma unroll
            for (int ab = 0; ab < 2; ++ab) {   // atom K-blocks (64 = 2 x 32)
                const _Float16* arow = ph + (i * 16 + nlane) * VPAD + ab * 32 + hiA;
                v16h af = ld16(arow, arow + 16);
                const _Float16* brow = vhT + (j * 16 + nlane) * VPAD + ab * 32 + hiA;
                v16h bf = ld16(brow, brow + 16);
                acc = wmma_f16(af, bf, acc);
            }
#pragma unroll
            for (int r = 0; r < 8; ++r) {
                int q = i * 16 + r + hiA;
                colsum += (q < len) ? acc[r] : 0.0f;
            }
        }
        colsum += __shfl_xor(colsum, 16, 32);  // combine lane halves (same e)
        if (lane < 16)
            out[(size_t)b * DD + j * 16 + lane] = colsum / (float)len;
    }
}

// ---------------------------------------------------------------------------
extern "C" void kernel_launch(void* const* d_in, const int* in_sizes, int n_in,
                              void* d_out, int out_size, void* d_ws, size_t ws_size,
                              hipStream_t stream) {
    const float* x       = (const float*)d_in[0];
    const int*   lengths = (const int*)  d_in[1];
    const float* Wq      = (const float*)d_in[2];
    const float* bq      = (const float*)d_in[3];
    const float* Wk      = (const float*)d_in[4];
    const float* bk      = (const float*)d_in[5];
    const float* Wv      = (const float*)d_in[6];
    const float* bv      = (const float*)d_in[7];
    float*       out     = (float*)d_out;
    _Float16*    wT      = (_Float16*)d_ws;     // 3*512*512 f16 = 1.5 MB

    // Opt in to >64KB dynamic LDS (no-op if already allowed).
    (void)hipFuncSetAttribute((const void*)atom_attention_kernel,
                              hipFuncAttributeMaxDynamicSharedMemorySize,
                              SMEM_BYTES);

    prep_weights<<<(3 * DD * DD + 255) / 256, 256, 0, stream>>>(Wq, Wk, Wv, wT);
    atom_attention_kernel<<<NB, NT, SMEM_BYTES, stream>>>(
        x, lengths, bq, bk, bv, wT, out);
}
